// My_Model_81243601371714
// MI455X (gfx1250) — matrix operand order repaired and checked by reading, not compiled, not run on hardware
//
#include <hip/hip_runtime.h>

// ---------------- constants ----------------
#define LDIM     8192          // L
#define TOUT     128           // outputs per block
#define NPOS     160           // interior positions computed per block (10 M-tiles)
#define YN       164           // Prim/y tile incl. halo
#define NTHREADS 256           // 8 wave32s

typedef __attribute__((ext_vector_type(16))) _Float16     v16h;
typedef __attribute__((ext_vector_type(8)))  float        v8f;
typedef __attribute__((ext_vector_type(4)))  unsigned int v4u;

union HV { v16h h; v4u u[2]; };

__device__ __forceinline__ float fast_tanh(float x) {
    // tanh(x) = 1 - 2/(exp(2x)+1)   (v_exp_f32 + v_rcp_f32)
    float e = __expf(2.0f * x);
    return 1.0f - 2.0f * __builtin_amdgcn_rcpf(e + 1.0f);
}

__device__ __forceinline__ float phi_of(float gm1, float g) {
    float aa = (g < 0.0f) ? -1.0f : 1.0f;       // min(sign(g),0)*2+1
    float ri = gm1 / (fmaxf(fabsf(g), 1e-15f) * aa);
    return (ri * ri + ri) / (ri * ri + 1.0f);
}

// ---------------- kernel 0: weight repack to f16 (L2-resident) ----------------
// w2t[co][k*64+ci]  (64 x 320),  w3t[n][k*64+ci]  (16 x 320, rows 4..15 = 0)
__global__ __launch_bounds__(256) void prep_weights(
    const float* __restrict__ W2, const float* __restrict__ W3,
    _Float16* __restrict__ w2t, _Float16* __restrict__ w3t)
{
    int idx = blockIdx.x * 256 + threadIdx.x;
    if (idx < 64 * 320) {
        int co = idx / 320, kk = idx % 320;
        int k = kk >> 6, ci = kk & 63;
        w2t[idx] = (_Float16)W2[(k * 64 + ci) * 64 + co];
    }
    if (idx < 16 * 320) {
        int n = idx / 320, kk = idx % 320;
        int k = kk >> 6, ci = kk & 63;
        float v = (n < 4) ? W3[(k * 64 + ci) * 4 + n] : 0.0f;
        w3t[idx] = (_Float16)v;
    }
}

// ---------------- kernel 1: per-row min / scale ----------------
__global__ __launch_bounds__(256) void row_minmax(
    const float* __restrict__ prim, float* __restrict__ mnsc)
{
    __shared__ float smx[256], smn[256];
    const int row = blockIdx.x, tid = threadIdx.x;
    const float* p = prim + (size_t)row * LDIM;
    float mx = -3.402823e38f, mn = 3.402823e38f;
    for (int j = tid; j < LDIM; j += 256) {
        float v = p[j];
        mx = fmaxf(mx, v); mn = fminf(mn, v);
    }
    smx[tid] = mx; smn[tid] = mn; __syncthreads();
    for (int s = 128; s > 0; s >>= 1) {
        if (tid < s) {
            smx[tid] = fmaxf(smx[tid], smx[tid + s]);
            smn[tid] = fminf(smn[tid], smn[tid + s]);
        }
        __syncthreads();
    }
    if (tid == 0) {
        float m = smn[0];
        mnsc[row * 2 + 0] = m;
        mnsc[row * 2 + 1] = fmaxf((smx[0] - m) * 0.5f, 1e-4f);
    }
}

// ---------------- kernel 2: fused conv stack (WMMA) + stencil + TVD flux ----------------
__global__ __launch_bounds__(NTHREADS) void fused_stencil(
    const float* __restrict__ prim,
    const float* __restrict__ W1, const float* __restrict__ b1,
    const float* __restrict__ b2, const float* __restrict__ b3,
    const _Float16* __restrict__ w2t, const _Float16* __restrict__ w3t,
    const float* __restrict__ mnsc, float* __restrict__ out)
{
    __shared__ float yv[YN], pv[YN];
    __shared__ float w1s[5 * 64], b1s[64], b2s[64], b3s[16];
    __shared__ __align__(16) _Float16 h1[NPOS * 64];
    __shared__ __align__(16) _Float16 h2[NPOS * 64];
    __shared__ float coef[NPOS * 4];
    __shared__ float grad[NPOS];
    __shared__ float flux[NPOS];

    const int tid   = threadIdx.x;
    const int row   = blockIdx.x >> 6;            // 64 tiles per row
    const int tile0 = (blockIdx.x & 63) * TOUT;

    const float mn = mnsc[row * 2 + 0];
    const float sc = mnsc[row * 2 + 1];
    const float inv_sc = __builtin_amdgcn_rcpf(sc);

    // ---- stage Prim tile (periodic) + small weights ----
    const float* prow = prim + (size_t)row * LDIM;
    for (int j = tid; j < YN; j += NTHREADS) {
        int gp = (tile0 - 12 + j + LDIM) & (LDIM - 1);
        float p = prow[gp];
        pv[j] = p;
        yv[j] = (p - mn) * inv_sc - 1.0f;
    }
    for (int x = tid; x < 5 * 64; x += NTHREADS) w1s[x] = W1[x];
    if (tid < 64) { b1s[tid] = b1[tid]; b2s[tid] = b2[tid]; }
    if (tid < 16) b3s[tid] = (tid < 4) ? b3[tid] : 0.0f;
    __syncthreads();

    // ---- layer 1: 1 -> 64 channels, K=5, tanh (VALU) ----
    for (int x = tid; x < NPOS * 64; x += NTHREADS) {
        int i = x >> 6, c = x & 63;
        float s = b1s[c];
        #pragma unroll
        for (int k = 0; k < 5; ++k) s = fmaf(yv[i + k], w1s[k * 64 + c], s);
        h1[x] = (_Float16)fast_tanh(s);
    }
    __syncthreads();

    const int wave = tid >> 5;
    const int lane = tid & 31;
    const int mrow = lane & 15;      // M (A) / N (B,C,D) index
    const int grp  = lane >> 4;      // lane half-group

    // ---- layer 2: 64 -> 64, K-dim = 5*64 = 320 via 10x K32 WMMA steps ----
    // Each wave owns ONE fixed N-tile and half the M-tiles: 10 B-fragments are
    // loaded once (80 VGPRs, intentional) and reused across 5 M-tiles.
    {
        const int ntile = wave & 3;      // fixed N-tile (output channels n*16..n*16+15)
        const int tgrp  = wave >> 2;     // M-tile parity (0 or 1)

        HV bf[10];
        #pragma unroll
        for (int s = 0; s < 10; ++s) {   // B 32x16 layout: lanes<16 K0-7/16-23, lanes>=16 K8-15/24-31
            const _Float16* bp = w2t + (ntile * 16 + mrow) * 320 + s * 32 + grp * 8;
            bf[s].u[0] = *(const v4u*)bp;
            bf[s].u[1] = *(const v4u*)(bp + 16);
        }
        const float bv = b2s[ntile * 16 + mrow];

        #pragma unroll 1
        for (int t = tgrp; t < 10; t += 2) {
            const int tbase = t * 16;
            v8f acc = (v8f){bv, bv, bv, bv, bv, bv, bv, bv};
            #pragma unroll
            for (int s = 0; s < 10; ++s) {
                const int k   = s >> 1;
                const int cih = (s & 1) * 32;
                int pi = tbase + mrow + k - 2;
                pi = pi < 0 ? 0 : (pi > NPOS - 1 ? NPOS - 1 : pi);
                HV a;   // A 16x32 layout: lanes<16 K0-7/16-23, lanes>=16 K8-15/24-31
                a.u[0] = *(const v4u*)&h1[pi * 64 + cih + grp * 8];
                a.u[1] = *(const v4u*)&h1[pi * 64 + cih + 16 + grp * 8];
                acc = __builtin_amdgcn_wmma_f32_16x16x32_f16(
                    false, a.h, false, bf[s].h, (short)0, acc, false, false);
            }
            #pragma unroll
            for (int r = 0; r < 8; ++r) {
                int po = tbase + r + grp * 8;    // C/D: VGPR r -> M = r + 8*grp
                h2[po * 64 + ntile * 16 + mrow] = (_Float16)fast_tanh(acc[r]);
            }
        }
    }
    __syncthreads();

    // ---- layer 3: 64 -> 4 (padded to 16 cols), one N-tile ----
    {
        HV bf[10];
        #pragma unroll
        for (int s = 0; s < 10; ++s) {
            const _Float16* bp = w3t + mrow * 320 + s * 32 + grp * 8;
            bf[s].u[0] = *(const v4u*)bp;
            bf[s].u[1] = *(const v4u*)(bp + 16);
        }
        const float bv = b3s[mrow];

        #pragma unroll 1
        for (int t = wave; t < 10; t += 8) {
            const int tbase = t * 16;
            v8f acc = (v8f){bv, bv, bv, bv, bv, bv, bv, bv};
            #pragma unroll
            for (int s = 0; s < 10; ++s) {
                const int k   = s >> 1;
                const int cih = (s & 1) * 32;
                int pi = tbase + mrow + k - 2;
                pi = pi < 0 ? 0 : (pi > NPOS - 1 ? NPOS - 1 : pi);
                HV a;
                a.u[0] = *(const v4u*)&h2[pi * 64 + cih + grp * 8];
                a.u[1] = *(const v4u*)&h2[pi * 64 + cih + 16 + grp * 8];
                acc = __builtin_amdgcn_wmma_f32_16x16x32_f16(
                    false, a.h, false, bf[s].h, (short)0, acc, false, false);
            }
            if (mrow < 4) {
                #pragma unroll
                for (int r = 0; r < 8; ++r) {
                    int po = tbase + r + grp * 8;
                    coef[po * 4 + mrow] = acc[r];
                }
            }
        }
    }
    __syncthreads();

    // ---- polynomial-accuracy layer + learned derivative ----
    {
        const float AP[5]    = {0.08333333f, -0.66666667f, 0.0f, 0.66666667f, -0.08333333f};
        const float NB0[5]   = {0.44721360f,  0.63819660f, -0.36180340f, -0.36180340f, -0.36180340f};
        const float NB1[5]   = {0.44721360f, -0.36180340f,  0.63819660f, -0.36180340f, -0.36180340f};
        const float NB2[5]   = {0.44721360f, -0.36180340f, -0.36180340f,  0.63819660f, -0.36180340f};
        const float NB3[5]   = {0.44721360f, -0.36180340f, -0.36180340f, -0.36180340f,  0.63819660f};
        for (int i = tid; i < NPOS; i += NTHREADS) {
            float c0 = coef[i * 4 + 0], c1 = coef[i * 4 + 1];
            float c2 = coef[i * 4 + 2], c3 = coef[i * 4 + 3];
            float g = 0.0f;
            #pragma unroll
            for (int k = 0; k < 5; ++k) {
                float al = AP[k] + c0 * NB0[k] + c1 * NB1[k] + c2 * NB2[k] + c3 * NB3[k];
                g = fmaf(al, pv[i + k], g);
            }
            grad[i] = g * 64.0f;   // / DX
        }
    }
    __syncthreads();

    // ---- TVD limiter + Godunov-type flux ----
    if (tid >= 1 && tid < NPOS - 1) {
        const int i = tid;
        float gi = grad[i], gm = grad[i - 1], gp = grad[i + 1];
        float ph_i = phi_of(gm, gi);
        float ph_p = phi_of(gi, gp);
        const float hdx = 0.5f * (1.0f / 64.0f);
        float uL = pv[i + 2] + hdx * ph_i * gi;
        float uR = pv[i + 3] - hdx * ph_p * gp;
        flux[i] = 0.25f * (uL * uL + uR * uR) - 0.25f * fabsf(uL + uR) * (uR - uL);
    }
    __syncthreads();

    // ---- divergence + Euler update ----
    if (tid < TOUT) {
        const int i = tid + 10;     // position index within tile frame
        out[(size_t)row * LDIM + tile0 + tid] = pv[i + 2] - 0.01f * (flux[i] - flux[i - 1]);
    }
}

// ---------------- launch ----------------
extern "C" void kernel_launch(void* const* d_in, const int* in_sizes, int n_in,
                              void* d_out, int out_size, void* d_ws, size_t ws_size,
                              hipStream_t stream) {
    const float* Prim = (const float*)d_in[0];
    const float* W1   = (const float*)d_in[1];
    const float* b1   = (const float*)d_in[2];
    const float* W2   = (const float*)d_in[3];
    const float* b2   = (const float*)d_in[4];
    const float* W3   = (const float*)d_in[5];
    const float* b3   = (const float*)d_in[6];
    float* out = (float*)d_out;

    const int B = in_sizes[0] / LDIM;   // 256

    float*    mnsc = (float*)d_ws;                                   // B*2 floats
    _Float16* w2t  = (_Float16*)((char*)d_ws + 4096);                // 64*320 f16
    _Float16* w3t  = (_Float16*)((char*)d_ws + 4096 + 64 * 320 * 2); // 16*320 f16

    prep_weights<<<80, 256, 0, stream>>>(W2, W3, w2t, w3t);
    row_minmax<<<B, 256, 0, stream>>>(Prim, mnsc);
    fused_stencil<<<B * (LDIM / TOUT), NTHREADS, 0, stream>>>(
        Prim, W1, b1, b2, b3, w2t, w3t, mnsc, out);
}
